// GNNModel_61967788147057
// MI455X (gfx1250) — matrix operand order repaired and checked by reading, not compiled, not run on hardware
//
#include <hip/hip_runtime.h>
#include <hip/hip_bf16.h>

typedef __attribute__((ext_vector_type(16))) _Float16 v16h;
typedef __attribute__((ext_vector_type(8)))  _Float16 v8h;
typedef __attribute__((ext_vector_type(8)))  float    v8f;

#define BN_EPS 1e-5f

// ---------------- utility kernels ----------------

__global__ void zero_f(float* __restrict__ p, size_t n) {
    size_t i = (size_t)blockIdx.x * blockDim.x + threadIdx.x;
    if (i < n) p[i] = 0.0f;
}

__global__ void degree_kernel(const int* __restrict__ dst, float* __restrict__ deg, int E) {
    int e = blockIdx.x * blockDim.x + threadIdx.x;
    if (e < E) atomicAdd(&deg[dst[e]], 1.0f);
}

__global__ void dinv_kernel(float* __restrict__ deg, int N) {
    int i = blockIdx.x * blockDim.x + threadIdx.x;
    if (i < N) {
        float d = deg[i];
        deg[i] = (d > 0.0f) ? rsqrtf(d) : 0.0f;   // in-place deg -> d^{-1/2}
    }
}

__global__ void norm_kernel(const int* __restrict__ src, const int* __restrict__ dst,
                            const float* __restrict__ dinv, float* __restrict__ norm, int E) {
    int e = blockIdx.x * blockDim.x + threadIdx.x;
    if (e < E) norm[e] = dinv[src[e]] * dinv[dst[e]];
}

__global__ void cvt_f32_f16(const float* __restrict__ in, _Float16* __restrict__ out, size_t n) {
    size_t i = (size_t)blockIdx.x * blockDim.x + threadIdx.x;
    if (i < n) out[i] = (_Float16)in[i];
}

// convert [rows,cin] f32 -> [rows,cout] f16 with zero padding (for W3: 40 -> 48 cols)
__global__ void cvt_f32_f16_pad(const float* __restrict__ in, _Float16* __restrict__ out,
                                int rows, int cin, int cout) {
    int i = blockIdx.x * blockDim.x + threadIdx.x;
    if (i >= rows * cout) return;
    int r = i / cout, c = i - r * cout;
    out[i] = (_Float16)((c < cin) ? in[r * cin + c] : 0.0f);
}

// out = relu(bn(agg + bias)) converted to f16 ; D = 64 (power of two -> mask)
__global__ void bias_bn_relu_cvt(const float* __restrict__ agg, const float* __restrict__ bias,
                                 const float* __restrict__ gamma, const float* __restrict__ beta,
                                 const float* __restrict__ mean, const float* __restrict__ var,
                                 _Float16* __restrict__ out, size_t n) {
    size_t i = (size_t)blockIdx.x * blockDim.x + threadIdx.x;
    if (i >= n) return;
    int f = (int)(i & 63);
    float v = agg[i] + bias[f];
    v = (v - mean[f]) * rsqrtf(var[f] + BN_EPS) * gamma[f] + beta[f];
    out[i] = (_Float16)fmaxf(v, 0.0f);
}

__global__ void add_bias(float* __restrict__ out, const float* __restrict__ b, size_t n, int D) {
    size_t i = (size_t)blockIdx.x * blockDim.x + threadIdx.x;
    if (i < n) out[i] += b[(int)(i % D)];
}

// ---------------- WMMA GEMM: C[M,Ncols] = A[M,K](f16) x B[K,Ncols](f16) ----------------
// One wave per 16x16 tile. K stepped by 32 (v_wmma_f32_16x16x32_f16).
// Register layouts per CDNA5 ISA 7.12.2:
//   A 16x32 f16 : row = lane&15 ; slot j -> K = 8*(lane>>4) + j + (j>=8 ? 8 : 0)
//   B 32x16 f16 : N = lane&15 ; slot j -> K as for A
//   C 16x16 f32 : slot v  -> M = v + 8*(lane>>4) ; N = lane&15
__global__ void gemm_wmma_f16(const _Float16* __restrict__ A, const _Float16* __restrict__ B,
                              float* __restrict__ C, int M, int Ncols, int K, int totalTiles) {
    const int lane = threadIdx.x & 31;
    const int wave = threadIdx.x >> 5;
    const int tile = blockIdx.x * (blockDim.x >> 5) + wave;
    if (tile >= totalTiles) return;              // wave-uniform: EXEC stays all-ones for WMMA
    const int ntiles = Ncols >> 4;
    const int mt = tile / ntiles;
    const int nt = tile - mt * ntiles;

    const int kgrp = (lane >> 4) << 3;           // 0 or 8
    const _Float16* Arow = A + (size_t)(mt * 16 + (lane & 15)) * K;

    v8f c = {};
    for (int k0 = 0; k0 < K; k0 += 32) {
        v8h a0 = *(const v8h*)(Arow + k0 + kgrp);        // K = k0+kgrp .. +7
        v8h a1 = *(const v8h*)(Arow + k0 + kgrp + 16);   // K = k0+kgrp+16 .. +23
        const _Float16* Bp = B + (size_t)(k0 + kgrp) * Ncols + nt * 16 + (lane & 15);
        v8h b0, b1;
        #pragma unroll
        for (int j = 0; j < 8; ++j) {
            b0[j] = Bp[(size_t)j * Ncols];               // K = k0+kgrp+j
            b1[j] = Bp[(size_t)(j + 16) * Ncols];        // K = k0+kgrp+16+j
        }
        v16h a = __builtin_shufflevector(a0, a1, 0,1,2,3,4,5,6,7,8,9,10,11,12,13,14,15);
        v16h b = __builtin_shufflevector(b0, b1, 0,1,2,3,4,5,6,7,8,9,10,11,12,13,14,15);
        c = __builtin_amdgcn_wmma_f32_16x16x32_f16(false, a, false, b, (short)0, c, false, false);
    }

    float* Cp = C + (size_t)(mt * 16 + ((lane >> 4) << 3)) * Ncols + nt * 16 + (lane & 15);
    #pragma unroll
    for (int v = 0; v < 8; ++v)
        Cp[(size_t)v * Ncols] = c[v];            // lanes 0-15 / 16-31 write 16 consecutive floats
}

// ---------------- edge gather-scale-scatter (L2-atomic bound stage) ----------------
// one wave per edge, lanes stride the feature dim
__global__ void scatter_edges(const float* __restrict__ t, const int* __restrict__ src,
                              const int* __restrict__ dst, const float* __restrict__ norm,
                              float* __restrict__ out, int E, int D, int strideIn, int strideOut) {
    int e = blockIdx.x * (blockDim.x >> 5) + (threadIdx.x >> 5);
    if (e >= E) return;
    int lane = threadIdx.x & 31;
    int s = src[e], d = dst[e];
    float w = norm[e];
    const float* trow = t + (size_t)s * strideIn;
    float* orow = out + (size_t)d * strideOut;
    for (int f = lane; f < D; f += 32)
        atomicAdd(&orow[f], trow[f] * w);
}

// ---------------- launch ----------------

static inline int cdiv(long long a, long long b) { return (int)((a + b - 1) / b); }

extern "C" void kernel_launch(void* const* d_in, const int* in_sizes, int n_in,
                              void* d_out, int out_size, void* d_ws, size_t ws_size,
                              hipStream_t stream) {
    const float* x  = (const float*)d_in[0];
    const int*   ei = (const int*)d_in[1];
    const float* W1 = (const float*)d_in[2];
    const float* b1 = (const float*)d_in[3];
    const float* W2 = (const float*)d_in[4];
    const float* b2 = (const float*)d_in[5];
    const float* W3 = (const float*)d_in[6];
    const float* b3 = (const float*)d_in[7];
    const float* g1 = (const float*)d_in[8];
    const float* be1= (const float*)d_in[9];
    const float* m1 = (const float*)d_in[10];
    const float* v1 = (const float*)d_in[11];
    const float* g2 = (const float*)d_in[12];
    const float* be2= (const float*)d_in[13];
    const float* m2 = (const float*)d_in[14];
    const float* v2 = (const float*)d_in[15];

    const int DIN = 128, DH = 64, DO = 40, DOP = 48;   // DOP: padded layer-3 cols
    const int N = in_sizes[0] / DIN;                    // 100000
    const int E = in_sizes[1] / 2;                      // 1000000
    const int* src = ei;
    const int* dst = ei + E;
    float* out = (float*)d_out;

    // workspace layout (float units, 256B aligned chunks)
    float* wsf = (float*)d_ws;
    size_t o = 0;
    auto al = [](size_t v){ return (v + 63) & ~(size_t)63; };
    float* deg  = wsf + o; o = al(o + (size_t)N);                 // degree -> dinv (in place)
    float* norm = wsf + o; o = al(o + (size_t)E);                 // per-edge norm
    float* t    = wsf + o; o = al(o + (size_t)N * DH);            // GEMM output (also N x 48 for L3)
    float* agg  = wsf + o; o = al(o + (size_t)N * DH);            // scatter accumulator
    _Float16* h16 = (_Float16*)(wsf + o); o = al(o + (size_t)N * DH);  // N x 128 halves (x) / N x 64 (h)
    _Float16* w16 = (_Float16*)(wsf + o);                         // up to 128x64 halves

    const int B256 = 256;

    // ---- normalization coefficients ----
    zero_f<<<cdiv(N,B256), B256, 0, stream>>>(deg, (size_t)N);
    degree_kernel<<<cdiv(E,B256), B256, 0, stream>>>(dst, deg, E);
    dinv_kernel<<<cdiv(N,B256), B256, 0, stream>>>(deg, N);
    norm_kernel<<<cdiv(E,B256), B256, 0, stream>>>(src, dst, deg, norm, E);

    // ---- layer 1: h = relu(bn1(scatter(x@W1 * norm) + b1)) ----
    cvt_f32_f16<<<cdiv((long long)N*DIN,B256), B256, 0, stream>>>(x, h16, (size_t)N*DIN);
    cvt_f32_f16<<<cdiv(DIN*DH,B256), B256, 0, stream>>>(W1, w16, (size_t)DIN*DH);
    {
        int tiles = (N/16) * (DH/16);
        gemm_wmma_f16<<<cdiv(tiles,8), B256, 0, stream>>>(h16, w16, t, N, DH, DIN, tiles);
    }
    zero_f<<<cdiv((long long)N*DH,B256), B256, 0, stream>>>(agg, (size_t)N*DH);
    scatter_edges<<<cdiv(E,8), B256, 0, stream>>>(t, src, dst, norm, agg, E, DH, DH, DH);
    bias_bn_relu_cvt<<<cdiv((long long)N*DH,B256), B256, 0, stream>>>(agg, b1, g1, be1, m1, v1,
                                                                      h16, (size_t)N*DH);

    // ---- layer 2 ----
    cvt_f32_f16<<<cdiv(DH*DH,B256), B256, 0, stream>>>(W2, w16, (size_t)DH*DH);
    {
        int tiles = (N/16) * (DH/16);
        gemm_wmma_f16<<<cdiv(tiles,8), B256, 0, stream>>>(h16, w16, t, N, DH, DH, tiles);
    }
    zero_f<<<cdiv((long long)N*DH,B256), B256, 0, stream>>>(agg, (size_t)N*DH);
    scatter_edges<<<cdiv(E,8), B256, 0, stream>>>(t, src, dst, norm, agg, E, DH, DH, DH);
    bias_bn_relu_cvt<<<cdiv((long long)N*DH,B256), B256, 0, stream>>>(agg, b2, g2, be2, m2, v2,
                                                                      h16, (size_t)N*DH);

    // ---- layer 3: out = scatter(h@W3 * norm) + b3 (W3 zero-padded 40->48 cols) ----
    cvt_f32_f16_pad<<<cdiv(DH*DOP,B256), B256, 0, stream>>>(W3, w16, DH, DO, DOP);
    {
        int tiles = (N/16) * (DOP/16);
        gemm_wmma_f16<<<cdiv(tiles,8), B256, 0, stream>>>(h16, w16, t, N, DOP, DH, tiles);
    }
    zero_f<<<cdiv((long long)N*DO,B256), B256, 0, stream>>>(out, (size_t)N*DO);
    scatter_edges<<<cdiv(E,8), B256, 0, stream>>>(t, src, dst, norm, out, E, DO, DOP, DO);
    add_bias<<<cdiv((long long)N*DO,B256), B256, 0, stream>>>(out, b3, (size_t)N*DO, DO);
}
